// ProjConvReducedI2T_68324339744986
// MI455X (gfx1250) — compile-verified, matrix operand order checked by reading, not run on hardware
//
#include <hip/hip_runtime.h>
#include <math.h>

// ---------------------------------------------------------------------------
// ProjConvReducedI2T on gfx1250 (MI455X, wave32, WMMA).
// All contractions on v_wmma_f32_16x16x32_f16. Weights pre-converted to f16;
// activations stored (b, t, c) f16 with padded time stride so every WMMA
// fragment load is an unguarded, aligned 16-byte vector load.
// ---------------------------------------------------------------------------

typedef __attribute__((ext_vector_type(16))) _Float16 v16h;
typedef __attribute__((ext_vector_type(8)))  _Float16 v8h;
typedef __attribute__((ext_vector_type(8)))  float    v8f;

#define L_    1024
#define R_    128
#define BI_   32
#define BT_   128
#define T_    64
#define TXT_  300
#define CP_   320      // padded TXT
#define TS_   66       // time stride for (b,t,c) activation buffers
#define EPS_  1e-5f

// K index held in f16-pair slot v (0..7) of the A/B fragment for lane half h.
__device__ __forceinline__ int kbase_of(int v, int h) {
  return (v < 4) ? (h * 8 + 2 * v) : (16 + h * 8 + 2 * (v - 4));
}

// One 16x16 (f32) += 16x32 (f16) x 32x16 (f16) WMMA step.
template <typename FA, typename FB>
__device__ __forceinline__ v8f wmma_tile(v8f c, int lane, FA ldA, FB ldB) {
  const int mn = lane & 15;
  const int h  = lane >> 4;
  v16h a, b;
#pragma unroll
  for (int v = 0; v < 8; ++v) {
    const int kb = kbase_of(v, h);
    a[2 * v]     = ldA(mn, kb);
    a[2 * v + 1] = ldA(mn, kb + 1);
    b[2 * v]     = ldB(kb, mn);
    b[2 * v + 1] = ldB(kb + 1, mn);
  }
  return __builtin_amdgcn_wmma_f32_16x16x32_f16(false, a, false, b,
                                                (short)0, c, false, false);
}

// ---------------------------------------------------------------------------
// Preprocessing
// ---------------------------------------------------------------------------

// cap_embed (BT,64,300) f32 -> cap16in (BT,TS,320) f16 (pad rows/cols pre-zeroed)
__global__ void k_pad_cap(const float* __restrict__ src, _Float16* __restrict__ dst) {
  int idx = blockIdx.x * 256 + threadIdx.x;        // BT*64*300
  int b = idx / (64 * 300);
  int rem = idx - b * (64 * 300);
  int t = rem / 300, c = rem - t * 300;
  dst[((size_t)b * TS_ + t) * CP_ + c] = (_Float16)src[idx];
}

// W_rt (128,300) -> wrt16 (128,320) zero-padded
__global__ void k_pad_wrt(const float* __restrict__ src, _Float16* __restrict__ dst) {
  int idx = blockIdx.x * 256 + threadIdx.x;        // 128*320
  int o = idx / CP_, c = idx - o * CP_;
  dst[idx] = (c < TXT_) ? (_Float16)src[(size_t)o * TXT_ + c] : (_Float16)0.f;
}

// generic f32 -> f16
__global__ void k_cvt(const float* __restrict__ src, _Float16* __restrict__ dst, int n) {
  int idx = blockIdx.x * 256 + threadIdx.x;
  if (idx < n) dst[idx] = (_Float16)src[idx];
}

// ca_W (2,128,64,2) -> ca16 [j][o][k = s*64+i] f16
__global__ void k_ca_pack(const float* __restrict__ src, _Float16* __restrict__ dst) {
  int idx = blockIdx.x * 256 + threadIdx.x;        // 2*128*128
  int k = idx & 127;
  int s = k >> 6, i = k & 63;
  int jo = idx >> 7;
  dst[idx] = (_Float16)src[(size_t)jo * 128 + i * 2 + s];
}

// ---------------------------------------------------------------------------
// Stage 0: image side
// ---------------------------------------------------------------------------
__global__ void k_img_global(const float* __restrict__ ie, float* __restrict__ ig) {
  int idx = blockIdx.x * 256 + threadIdx.x;        // BI*L
  int i = idx >> 10, l = idx & (L_ - 1);
  float s = 0.f;
  for (int t = 0; t < 36; ++t) s += ie[((size_t)i * 36 + t) * L_ + l];
  ig[idx] = s * (1.0f / 36.0f);
}

__global__ void k_img_norm(const float* __restrict__ ig, float* __restrict__ nrm) {
  __shared__ float sm[256];
  int i = blockIdx.x;
  float s = 0.f;
  for (int l = threadIdx.x; l < L_; l += 256) { float v = ig[(size_t)i * L_ + l]; s += v * v; }
  sm[threadIdx.x] = s; __syncthreads();
  for (int st = 128; st > 0; st >>= 1) {
    if ((int)threadIdx.x < st) sm[threadIdx.x] += sm[threadIdx.x + st];
    __syncthreads();
  }
  if (!threadIdx.x) nrm[i] = sqrtf(sm[0]);
}

__global__ void k_img_vec(const float* __restrict__ W, const float* __restrict__ bb,
                          const float* __restrict__ ig, float* __restrict__ iv) {
  int idx = blockIdx.x * 256 + threadIdx.x;        // BI*R
  int i = idx >> 7, r = idx & (R_ - 1);
  const float* w = W + (size_t)r * L_;
  const float* g = ig + (size_t)i * L_;
  float s = bb[r];
  for (int c = 0; c < L_; ++c) s += w[c] * g[c];
  iv[idx] = s;
}

// ---------------------------------------------------------------------------
// cap_red: cap16[b,t,o] = sum_c wrt16[o,c]*cap16in[b,t,c] + b_rt[o]
// M=128(o), K=320, N=64(t). Unguarded, fully vectorizable loads.
// ---------------------------------------------------------------------------
__global__ void __launch_bounds__(256)
k_cap_red(const _Float16* __restrict__ Wrt, const float* __restrict__ brt,
          const _Float16* __restrict__ cin, _Float16* __restrict__ out) {
  const int b = blockIdx.x;
  const int lane = threadIdx.x & 31;
  const int wv = threadIdx.x >> 5;
  const int obase = wv * 16;
  const _Float16* xb = cin + (size_t)b * TS_ * CP_;
  for (int nt = 0; nt < 4; ++nt) {
    const int tb = nt * 16;
    v8f acc = {};
    for (int k0 = 0; k0 < CP_; k0 += 32) {
      acc = wmma_tile(acc, lane,
        [&](int m, int k) -> _Float16 { return Wrt[(size_t)(obase + m) * CP_ + k0 + k]; },
        [&](int k, int n) -> _Float16 { return xb[(size_t)(tb + n) * CP_ + k0 + k]; });
    }
    const int h = lane >> 4, n = lane & 15, t = tb + n;
    const int o0 = obase + 8 * h;
    v8h ov;
#pragma unroll
    for (int r = 0; r < 8; ++r) ov[r] = (_Float16)(acc[r] + brt[o0 + r]);
    *(v8h*)(out + ((size_t)b * TS_ + t) * R_ + o0) = ov;
  }
}

// ---------------------------------------------------------------------------
// Hypernet matvecs
// ---------------------------------------------------------------------------
__global__ void k_hyper(const float* __restrict__ mkW, const float* __restrict__ mkb,
                        const float* __restrict__ mbW, const float* __restrict__ mbb,
                        const float* __restrict__ base,
                        float* __restrict__ kern_raw, float* __restrict__ kbias) {
  int idx = blockIdx.x * 256 + threadIdx.x;        // 2*16384 + 256
  if (idx < 2 * 16384) {
    const float* w = mkW + (size_t)idx * 128;
    float s = mkb[idx];
    for (int c = 0; c < 128; ++c) s += w[c] * base[c];
    kern_raw[idx] = s;
  } else {
    int t = idx - 2 * 16384;
    const float* w = mbW + (size_t)t * 128;
    float s = mbb[t];
    for (int c = 0; c < 128; ++c) s += w[c] * base[c];
    kbias[t] = s;
  }
}

// Weight-norm BN over each out-channel's 128 weights; emit f16 in permuted
// layout kern16[jo][k = s*64 + i]  (thread tid = original i*2+s index).
__global__ void k_wn(const float* __restrict__ kern_raw,
                     const float* __restrict__ g, const float* __restrict__ bb,
                     _Float16* __restrict__ kern16) {
  __shared__ float s1[128], s2[128];
  int jo = blockIdx.x;                             // j*128 + o
  int e = threadIdx.x;                             // i*2 + s
  float v = kern_raw[(size_t)jo * 128 + e];
  s1[e] = v; s2[e] = v * v;
  __syncthreads();
  for (int st = 64; st > 0; st >>= 1) {
    if (e < st) { s1[e] += s1[e + st]; s2[e] += s2[e + st]; }
    __syncthreads();
  }
  float m  = s1[0] * (1.f / 128.f);
  float va = s2[0] * (1.f / 128.f) - m * m;
  float rs = rsqrtf(va + EPS_);
  int k = (e & 1) * 64 + (e >> 1);                 // s*64 + i
  kern16[(size_t)jo * 128 + k] = (_Float16)(g[jo] * (v - m) * rs + bb[jo]);
}

// ---------------------------------------------------------------------------
// Grouped conv as GEMM, K = s*64+i: h[b,t,chOff+o] = sum_k kw[o,k]*x[b,t+s,64g+i]
// x is (b, TS, 128) f16; h is (b, 64, 256) f32. Pad cols (t >= Tout) stored 0.
// ---------------------------------------------------------------------------
__global__ void __launch_bounds__(256)
k_gconv(const _Float16* __restrict__ x, int Tout,
        const _Float16* __restrict__ kw, const float* __restrict__ kb,
        float* __restrict__ h, int chOff) {
  const int b = blockIdx.x;
  const int lane = threadIdx.x & 31;
  const int wv = threadIdx.x >> 5;
  const int obase = wv * 16;
  const int g = obase >> 6;
  const _Float16* xb = x + (size_t)b * TS_ * R_ + 64 * g;
  for (int nt = 0; nt < 4; ++nt) {
    const int tb = nt * 16;
    v8f acc = {};
    for (int k0 = 0; k0 < 128; k0 += 32) {
      acc = wmma_tile(acc, lane,
        [&](int m, int k) -> _Float16 { return kw[(size_t)(obase + m) * 128 + k0 + k]; },
        [&](int k, int n) -> _Float16 {
          int kk = k0 + k, s = kk >> 6, i = kk & 63;
          return xb[(size_t)(tb + n + s) * R_ + i];
        });
    }
    const int hh = lane >> 4, n = lane & 15, t = tb + n;
    const int o0 = chOff + obase + 8 * hh;
    const bool valid = (t < Tout);
    float* hp = h + ((size_t)b * 64 + t) * 256 + o0;
#pragma unroll
    for (int r = 0; r < 8; ++r)
      hp[r] = valid ? (acc[r] + kb[obase + 8 * hh + r]) : 0.f;
  }
}

// BN training stats per channel over (BT, Tout);  h is (b, 64, 256)
__global__ void k_bnstats(const float* __restrict__ h, int Tout,
                          float* __restrict__ mo, float* __restrict__ ro) {
  __shared__ float s1[256], s2[256];
  const int ch = blockIdx.x;
  const int N = BT_ * Tout;
  float a = 0.f, q = 0.f;
  for (int idx = threadIdx.x; idx < N; idx += 256) {
    int b = idx / Tout, t = idx - b * Tout;
    float v = h[((size_t)b * 64 + t) * 256 + ch];
    a += v; q += v * v;
  }
  s1[threadIdx.x] = a; s2[threadIdx.x] = q;
  __syncthreads();
  for (int st = 128; st > 0; st >>= 1) {
    if ((int)threadIdx.x < st) {
      s1[threadIdx.x] += s1[threadIdx.x + st];
      s2[threadIdx.x] += s2[threadIdx.x + st];
    }
    __syncthreads();
  }
  if (!threadIdx.x) {
    float m  = s1[0] / (float)N;
    float va = s2[0] / (float)N - m * m;
    mo[ch] = m;
    ro[ch] = rsqrtf(va + EPS_);
  }
}

// 1x1 mixing conv, fused BN+ReLU on input: M=128(o), K=256(c), N=64(t)
// h (b,64,256) f32 -> out (b,TS,128) f16, pad cols zeroed.
__global__ void __launch_bounds__(256)
k_c1(const float* __restrict__ h, int Tout,
     const float* __restrict__ bnm, const float* __restrict__ bnr,
     const float* __restrict__ bng, const float* __restrict__ bnb,
     const _Float16* __restrict__ c1W, const float* __restrict__ c1b,
     _Float16* __restrict__ out) {
  const int b = blockIdx.x;
  const int lane = threadIdx.x & 31;
  const int wv = threadIdx.x >> 5;
  const int obase = wv * 16;
  const float* hb = h + (size_t)b * 64 * 256;
  for (int nt = 0; nt < 4; ++nt) {
    const int tb = nt * 16;
    v8f acc = {};
    for (int k0 = 0; k0 < 256; k0 += 32) {
      acc = wmma_tile(acc, lane,
        [&](int m, int k) -> _Float16 { return c1W[(size_t)(obase + m) * 256 + k0 + k]; },
        [&](int k, int n) -> _Float16 {
          int c = k0 + k;
          float v = hb[(size_t)(tb + n) * 256 + c];
          v = bng[c] * (v - bnm[c]) * bnr[c] + bnb[c];
          return (_Float16)(v > 0.f ? v : 0.f);
        });
    }
    const int hh = lane >> 4, n = lane & 15, t = tb + n;
    const int o0 = obase + 8 * hh;
    const bool valid = (t < Tout);
    v8h ov;
#pragma unroll
    for (int r = 0; r < 8; ++r)
      ov[r] = valid ? (_Float16)(acc[r] + c1b[o0 + r]) : (_Float16)0.f;
    *(v8h*)(out + ((size_t)b * TS_ + t) * R_ + o0) = ov;
  }
}

// txt_fc + max over t: M=1024(l), K=128(c), N=64; x2 pad cols are zero so only
// a register-level -inf mask is needed for t >= Tout.
__global__ void __launch_bounds__(256)
k_txtfc_max(const _Float16* __restrict__ x, int Tout,
            const _Float16* __restrict__ Wfc, const float* __restrict__ bfc,
            float* __restrict__ maxx) {
  const int b = blockIdx.x;
  const int lane = threadIdx.x & 31;
  const int wv = threadIdx.x >> 5;
  const int lbase = ((int)blockIdx.y * 8 + wv) * 16;
  const _Float16* xb = x + (size_t)b * TS_ * R_;
  v8f acc[4];
  for (int nt = 0; nt < 4; ++nt) {
    const int tb = nt * 16;
    v8f a = {};
    for (int k0 = 0; k0 < 128; k0 += 32) {
      a = wmma_tile(a, lane,
        [&](int m, int k) -> _Float16 { return Wfc[(size_t)(lbase + m) * R_ + k0 + k]; },
        [&](int k, int n) -> _Float16 { return xb[(size_t)(tb + n) * R_ + k0 + k]; });
    }
    acc[nt] = a;
  }
  const int hh = lane >> 4, n = lane & 15;
#pragma unroll
  for (int r = 0; r < 8; ++r) {
    float m = -3.0e38f;
#pragma unroll
    for (int nt = 0; nt < 4; ++nt) {
      int t = nt * 16 + n;
      if (t < Tout) m = fmaxf(m, acc[nt][r]);
    }
    for (int off = 1; off < 16; off <<= 1)
      m = fmaxf(m, __shfl_xor(m, off, 32));
    if (n == 0) {
      int l = lbase + r + 8 * hh;
      maxx[(size_t)b * L_ + l] = m + bfc[l];
    }
  }
}

// sims[i,b] = dot(maxx[b], ig[i]) / (||maxx[b]|| * ||ig[i]||)
__global__ void k_final(const float* __restrict__ maxx, const float* __restrict__ ig,
                        const float* __restrict__ inorm, int i, float* __restrict__ out) {
  __shared__ float s1[256], s2[256];
  const int b = blockIdx.x;
  float d = 0.f, q = 0.f;
  for (int l = threadIdx.x; l < L_; l += 256) {
    float v = maxx[(size_t)b * L_ + l];
    d += v * ig[(size_t)i * L_ + l];
    q += v * v;
  }
  s1[threadIdx.x] = d; s2[threadIdx.x] = q;
  __syncthreads();
  for (int st = 128; st > 0; st >>= 1) {
    if ((int)threadIdx.x < st) {
      s1[threadIdx.x] += s1[threadIdx.x + st];
      s2[threadIdx.x] += s2[threadIdx.x + st];
    }
    __syncthreads();
  }
  if (!threadIdx.x) out[(size_t)i * BT_ + b] = s1[0] / (sqrtf(s2[0]) * inorm[i]);
}

// ---------------------------------------------------------------------------
extern "C" void kernel_launch(void* const* d_in, const int* in_sizes, int n_in,
                              void* d_out, int out_size, void* d_ws, size_t ws_size,
                              hipStream_t stream) {
  (void)in_sizes; (void)n_in; (void)out_size; (void)ws_size;
  const float* img_embed = (const float*)d_in[0];
  const float* cap_embed = (const float*)d_in[1];
  const float* W_ri = (const float*)d_in[2];
  const float* b_ri = (const float*)d_in[3];
  const float* W_rt = (const float*)d_in[4];
  const float* b_rt = (const float*)d_in[5];
  const float* mk_W = (const float*)d_in[6];
  const float* mk_b = (const float*)d_in[7];
  const float* mbias_W = (const float*)d_in[8];
  const float* mbias_b = (const float*)d_in[9];
  const float* wn_g = (const float*)d_in[10];
  const float* wn_b = (const float*)d_in[11];
  const float* ca_W = (const float*)d_in[12];
  const float* ca_b = (const float*)d_in[13];
  const float* bn_g = (const float*)d_in[14];
  const float* bn_b = (const float*)d_in[15];
  const float* c1_W = (const float*)d_in[16];
  const float* c1_b = (const float*)d_in[17];
  const float* W_fc = (const float*)d_in[18];
  const float* b_fc = (const float*)d_in[19];
  float* out = (float*)d_out;

  // ---- workspace layout ----
  float* p = (float*)d_ws;
  float* f_ig    = p; p += BI_ * L_;
  float* f_inorm = p; p += BI_;
  float* f_iv    = p; p += BI_ * R_;
  float* f_kraw  = p; p += 2 * 16384;
  float* f_kb    = p; p += 256;
  float* f_h     = p; p += (size_t)BT_ * 64 * 256;       // 8 MB
  float* f_bnm   = p; p += 256;
  float* f_bnr   = p; p += 256;
  float* f_max   = p; p += (size_t)BT_ * L_;

  _Float16* hp = (_Float16*)p;
  // zeroed-every-call region: [cap16in, cap16, x1, x2]
  _Float16* f_cin  = hp; hp += (size_t)BT_ * TS_ * CP_;  // padded caption f16
  _Float16* f_cap  = hp; hp += (size_t)BT_ * TS_ * R_;   // cap_red out
  _Float16* f_x1   = hp; hp += (size_t)BT_ * TS_ * R_;
  _Float16* f_x2   = hp; hp += (size_t)BT_ * TS_ * R_;
  size_t zero_halves = (size_t)(hp - f_cin);
  _Float16* f_wrt  = hp; hp += 128 * CP_;
  _Float16* f_wfc  = hp; hp += L_ * R_;
  _Float16* f_c1w  = hp; hp += 2 * 128 * 256;
  _Float16* f_ca   = hp; hp += 2 * 16384;
  _Float16* f_kern = hp; hp += 2 * 16384;

  // ---- one-time-per-call preprocessing ----
  hipMemsetAsync(f_cin, 0, zero_halves * sizeof(_Float16), stream);
  k_pad_cap<<<(BT_ * 64 * 300) / 256, 256, 0, stream>>>(cap_embed, f_cin);
  k_pad_wrt<<<(128 * CP_) / 256, 256, 0, stream>>>(W_rt, f_wrt);
  k_cvt<<<(L_ * R_) / 256, 256, 0, stream>>>(W_fc, f_wfc, L_ * R_);
  k_cvt<<<(2 * 128 * 256) / 256, 256, 0, stream>>>(c1_W, f_c1w, 2 * 128 * 256);
  k_ca_pack<<<(2 * 16384) / 256, 256, 0, stream>>>(ca_W, f_ca);

  k_img_global<<<(BI_ * L_) / 256, 256, 0, stream>>>(img_embed, f_ig);
  k_img_norm<<<BI_, 256, 0, stream>>>(f_ig, f_inorm);
  k_img_vec<<<(BI_ * R_) / 256, 256, 0, stream>>>(W_ri, b_ri, f_ig, f_iv);
  k_cap_red<<<BT_, 256, 0, stream>>>(f_wrt, b_rt, f_cin, f_cap);

  for (int i = 0; i < BI_; ++i) {
    const float* base = f_iv + (size_t)i * R_;
    k_hyper<<<(2 * 16384 + 256) / 256, 256, 0, stream>>>(
        mk_W, mk_b, mbias_W, mbias_b, base, f_kraw, f_kb);
    k_wn<<<256, 128, 0, stream>>>(f_kraw, wn_g, wn_b, f_kern);

    // Mixed block j=0: input cap16 (valid t<64 -> Tout=63)
    k_gconv<<<BT_, 256, 0, stream>>>(f_cap, 63, f_kern, f_kb, f_h, 0);
    k_gconv<<<BT_, 256, 0, stream>>>(f_cap, 63, f_ca, ca_b, f_h, 128);
    k_bnstats<<<256, 256, 0, stream>>>(f_h, 63, f_bnm, f_bnr);
    k_c1<<<BT_, 256, 0, stream>>>(f_h, 63, f_bnm, f_bnr, bn_g, bn_b, f_c1w, c1_b, f_x1);

    // Mixed block j=1: input x1 (valid t<63 -> Tout=62)
    k_gconv<<<BT_, 256, 0, stream>>>(f_x1, 62, f_kern + 16384, f_kb + 128, f_h, 0);
    k_gconv<<<BT_, 256, 0, stream>>>(f_x1, 62, f_ca + 16384, ca_b + 128, f_h, 128);
    k_bnstats<<<256, 256, 0, stream>>>(f_h, 62, f_bnm, f_bnr);
    k_c1<<<BT_, 256, 0, stream>>>(f_h, 62, f_bnm, f_bnr, bn_g + 256, bn_b + 256,
                                  f_c1w + 32768, c1_b + 128, f_x2);

    // txt_fc + max over t, fused cosine similarity vs image i
    k_txtfc_max<<<dim3(BT_, 8), 256, 0, stream>>>(f_x2, 62, f_wfc, b_fc, f_max);
    k_final<<<BT_, 256, 0, stream>>>(f_max, f_ig, f_inorm, i, out);
  }
}
